// GemliteLinear_90409061580808
// MI455X (gfx1250) — matrix operand order repaired and checked by reading, not compiled, not run on hardware
//
#include <hip/hip_runtime.h>

typedef _Float16 h2   __attribute__((ext_vector_type(2)));
typedef _Float16 v8h  __attribute__((ext_vector_type(8)));
typedef _Float16 v16h __attribute__((ext_vector_type(16)));
typedef float    v4f  __attribute__((ext_vector_type(4)));
typedef float    v8f  __attribute__((ext_vector_type(8)));

#define K_DIM   4096
#define N_DIM   14336
#define GROUPS  32          // group_size = 128 => 4 K-chunks of 32 per group
#define KC      (K_DIM/32)  // 128 K-chunks
#define WPB     4           // waves per block
#define THREADS (WPB*32)

__global__ __launch_bounds__(THREADS)
void w3a16_wmma_gemm(const float* __restrict__ x,
                     const int*   __restrict__ w1,   // (K/32, N) 1-bit plane
                     const int*   __restrict__ w2,   // (K/16, N) 2-bit plane
                     const float* __restrict__ scales,
                     const float* __restrict__ zeros,
                     float*       __restrict__ out)
{
    // A fragments for the whole K range, in contiguous k order (half s = k 16*hi + s), same order as B:
    // fragment f = kc*32 + lane, 16 halves each -> 128 KB (WGP LDS is 320 KB)
    __shared__ _Float16 lds_a[KC * 32 * 16];

    const int tid  = threadIdx.x;
    const int lane = tid & 31;
    const int wave = tid >> 5;

    // ---- Stage x (f32) -> f16 A fragments in LDS ----
    // Lane l of fragment kc: M = l&15, hi = l>>4
    //   halves[0..7]  = x[M][kc*32 + hi*16 +     j]
    //   halves[8..15] = x[M][kc*32 + hi*16 + 8 + j]
    for (int f = tid; f < KC * 32; f += THREADS) {
        const int kc = f >> 5;
        const int l  = f & 31;
        const int m  = l & 15;
        const int kb = kc * 32 + (l >> 4) * 16;
        const float* src = x + m * K_DIM + kb;
        v4f a0 = *(const v4f*)(src);
        v4f a1 = *(const v4f*)(src + 4);
        v4f a2 = *(const v4f*)(src + 8);
        v4f a3 = *(const v4f*)(src + 12);
        v8h lo = { (_Float16)a0[0], (_Float16)a0[1], (_Float16)a0[2], (_Float16)a0[3],
                   (_Float16)a1[0], (_Float16)a1[1], (_Float16)a1[2], (_Float16)a1[3] };
        v8h hi = { (_Float16)a2[0], (_Float16)a2[1], (_Float16)a2[2], (_Float16)a2[3],
                   (_Float16)a3[0], (_Float16)a3[1], (_Float16)a3[2], (_Float16)a3[3] };
        v8h* dst = (v8h*)(lds_a + (size_t)f * 16);
        dst[0] = lo;   // ds_store_b128
        dst[1] = hi;   // ds_store_b128
    }
    __syncthreads();

    // ---- Per-wave 16x16 output tile over full K ----
    const int tileN = blockIdx.x * WPB + wave;
    const int col   = tileN * 16 + (lane & 15);
    const int kh    = lane >> 4;              // which 16-K half of each 32-K chunk
    const int sh1   = kh * 16;

    const int*   p1 = w1 + col;               // row kc       (stride N)
    const int*   p2 = w2 + kh * N_DIM + col;  // row 2*kc+kh  (stride 2N)
    const float* ps = scales + col;
    const float* pz = zeros  + col;

    const _Float16* afrag = lds_a + (size_t)lane * 16;

    v8f acc = {};
    const h2 kM1024 = { (_Float16)(-1024.0f), (_Float16)(-1024.0f) };

    // --- group-ahead software pipeline: weights are streamed (read-once, NT) ---
    auto load_group = [&](int g, unsigned b1w[4], unsigned b2w[4],
                          float& s, float& z) {
#pragma unroll
        for (int c = 0; c < 4; ++c) {
            const int kc = g * 4 + c;
            b1w[c] = (unsigned)__builtin_nontemporal_load(p1 + (size_t)kc * N_DIM);
            b2w[c] = (unsigned)__builtin_nontemporal_load(p2 + (size_t)(2 * kc) * N_DIM);
        }
        s = ps[(size_t)g * N_DIM];
        z = pz[(size_t)g * N_DIM];
    };

    auto compute_group = [&](int g, const unsigned b1w[4], const unsigned b2w[4],
                             float s, float z) {
        const _Float16 shh = (_Float16)s;
        const _Float16 bhh = (_Float16)(-z * s);
        const h2 s2    = { shh, shh };
        const h2 bias2 = { bhh, bhh };
#pragma unroll
        for (int c = 0; c < 4; ++c) {
            const int kc = g * 4 + c;
            const unsigned bits1 = (b1w[c] >> sh1) & 0xFFFFu;
            const unsigned bits2 = b2w[c];

            // Dequant 16 x 3-bit values -> f16 via 0x6400|q bit trick + pk math
            v16h b;
#pragma unroll
            for (int j = 0; j < 8; ++j) {
                const unsigned q2  = bits2 >> (4 * j);
                const unsigned q1  = bits1 >> (2 * j);
                const unsigned qlo = ((q2 & 3u) << 1) | (q1 & 1u);
                const unsigned qhi = (((q2 >> 2) & 3u) << 1) | ((q1 >> 1) & 1u);
                const unsigned word = 0x64006400u | qlo | (qhi << 16);
                h2 t = __builtin_bit_cast(h2, word); // (1024+qlo, 1024+qhi) exact
                t = t + kM1024;                      // v_pk_add_f16 -> (qlo, qhi)
                h2 wv = t * s2 + bias2;              // v_pk_fma_f16: (q - z)*s
                b[2 * j]     = wv[0];
                b[2 * j + 1] = wv[1];
            }

            // A fragment from LDS (2x ds_load_b128)
            const v16h a = *(const v16h*)(afrag + (size_t)kc * 32 * 16);

            acc = __builtin_amdgcn_wmma_f32_16x16x32_f16(
                false, a, false, b, (short)0, acc, false, false);
        }
    };

    unsigned b1A[4], b2A[4], b1B[4], b2B[4];
    float sA, zA, sB, zB;

    load_group(0, b1A, b2A, sA, zA);
    for (int g = 0; g < GROUPS; g += 2) {
        load_group(g + 1, b1B, b2B, sB, zB);   // prefetch while computing A
        compute_group(g, b1A, b2A, sA, zA);
        if (g + 2 < GROUPS)
            load_group(g + 2, b1A, b2A, sA, zA); // prefetch while computing B
        compute_group(g + 1, b1B, b2B, sB, zB);
    }

    // ---- Store C: lane 0-15 -> M=v, lane 16-31 -> M=8+v, N = col ----
    float* po = out + col;
#pragma unroll
    for (int v = 0; v < 8; ++v) {
        __builtin_nontemporal_store(acc[v], po + (size_t)(v + 8 * kh) * N_DIM);
    }
}

extern "C" void kernel_launch(void* const* d_in, const int* in_sizes, int n_in,
                              void* d_out, int out_size, void* d_ws, size_t ws_size,
                              hipStream_t stream) {
    const float* x      = (const float*)d_in[0];
    const int*   w1     = (const int*)  d_in[1];
    const int*   w2     = (const int*)  d_in[2];
    const float* scales = (const float*)d_in[3];
    const float* zeros  = (const float*)d_in[4];
    float*       out    = (float*)d_out;

    const int grid = N_DIM / (16 * WPB);   // 224 blocks x 128 threads (4 waves)
    w3a16_wmma_gemm<<<grid, THREADS, 0, stream>>>(x, w1, w2, scales, zeros, out);
}
